// MultiHeadLatentMoELayer_2877628088603
// MI455X (gfx1250) — matrix operand order repaired and checked by reading, not compiled, not run on hardware
//
#include <hip/hip_runtime.h>
#include <hip/hip_bf16.h>
#include <math.h>

// Problem sizes (fixed by the reference)
#define T_TOK 8192
#define DM    1024
#define NHH   8
#define DHH   128
#define NEXP  8
#define DHID  512

typedef __attribute__((ext_vector_type(16))) __bf16 v16bf;
typedef __attribute__((ext_vector_type(8)))  float  v8f;
typedef __attribute__((ext_vector_type(4)))  unsigned v4u_t;
typedef __attribute__((ext_vector_type(8)))  int      v8i_t;
typedef __attribute__((ext_vector_type(4)))  int      v4i_t;

// ---------------- WMMA fragment helpers (layouts per CDNA5 ISA) ---
// A 16x32 bf16 (MxK): lane m=l&15; lanes 0-15 hold K in {0..7,16..23},
// lanes 16-31 hold K in {8..15,24..31}.  Two 16B loads per lane.
__device__ __forceinline__ v16bf load_a_frag(const __bf16* base, int ld, int lane) {
  const int m    = lane & 15;
  const int koff = (lane >> 4) * 8;
  const __bf16* p = base + (size_t)m * ld + koff;
  union { v16bf v; uint4 q[2]; } u;
  u.q[0] = *reinterpret_cast<const uint4*>(p);       // K = koff..koff+7
  u.q[1] = *reinterpret_cast<const uint4*>(p + 16);  // K = koff+16..koff+23
  return u.v;
}

// B 32x16 bf16 (KxN) where memory holds W[n][k] row-major (B[k][n] = W[n*ld+k]):
// lane n=l&15; lanes 0-15 hold K in {0..7,16..23}, lanes 16-31 K in {8..15,24..31}.
__device__ __forceinline__ v16bf load_b_frag_wt(const __bf16* W, int ld, int lane) {
  const int n  = lane & 15;
  const int kb = (lane >> 4) * 8;
  const __bf16* p = W + (size_t)n * ld + kb;
  union { v16bf v; uint4 q[2]; } u;
  u.q[0] = *reinterpret_cast<const uint4*>(p);
  u.q[1] = *reinterpret_cast<const uint4*>(p + 16);
  return u.v;
}

__device__ __forceinline__ v8f wmma_bf16(v16bf a, v16bf b, v8f c) {
  return __builtin_amdgcn_wmma_f32_16x16x32_bf16(false, a, false, b, (short)0, c,
                                                 false, false);
}

// ---------------- TDM: async 2D tile -> LDS (CDNA5 ISA) ---
// 2D D# (count=1, type=2, data_size=2B, tensor dims == tile dims).  LDS dest
// is packed: row stride in LDS = rowElems * 2B.  Tracked by TENSORcnt.
__device__ __forceinline__ void tdm_load_2d_bf16(unsigned lds_off,
                                                 const void* gptr,
                                                 unsigned rowElems,
                                                 unsigned rows,
                                                 unsigned strideElems) {
  unsigned long long ga = (unsigned long long)gptr;
  v4u_t g0;
  g0[0] = 1u;                                   // count=1, no gather
  g0[1] = lds_off;                              // lds_addr
  g0[2] = (unsigned)ga;                         // global_addr[31:0]
  g0[3] = (unsigned)((ga >> 32) & 0x1ffffffull) // global_addr[56:32]
          | 0x80000000u;                        // type=2 ("image")
  // group1: [17:16]=data_size(1->2B), [79:48]=tensor_dim0, [111:80]=tensor_dim1,
  // [127:112]=tile_dim0, [143:128]=tile_dim1, [159:144]=tile_dim2(0),
  // [207:160]=tensor_dim0_stride, [255:208]=tensor_dim1_stride(0)
  unsigned long long q0 = (1ull << 16) |
                          ((unsigned long long)(rowElems & 0xffffu) << 48);
  unsigned long long q1 = (unsigned long long)(rowElems >> 16) |
                          ((unsigned long long)rows << 16) |
                          ((unsigned long long)(rowElems & 0xffffu) << 48);
  unsigned long long q2 = (unsigned long long)(rows & 0xffffu) |
                          ((unsigned long long)strideElems << 32);
  unsigned long long q3 = 0ull;
  v8i_t g1;
  g1[0] = (int)q0; g1[1] = (int)(q0 >> 32);
  g1[2] = (int)q1; g1[3] = (int)(q1 >> 32);
  g1[4] = (int)q2; g1[5] = (int)(q2 >> 32);
  g1[6] = (int)q3; g1[7] = (int)(q3 >> 32);
  v4i_t gz = {0, 0, 0, 0};
#if defined(__clang_major__) && __clang_major__ >= 23
  v8i_t gz8 = {0, 0, 0, 0, 0, 0, 0, 0};
  __builtin_amdgcn_tensor_load_to_lds(g0, g1, gz, gz, gz8, 0);
#else
  __builtin_amdgcn_tensor_load_to_lds(g0, g1, gz, gz, 0);
#endif
}

__device__ __forceinline__ unsigned lds_offset_of(const void* p) {
  return (unsigned)(unsigned long long)p;  // low 32 bits = LDS byte offset
}

// ---------------- elementwise casts ----------------------------------------
__global__ void cast_f32_bf16_kernel(const float* __restrict__ in,
                                     __bf16* __restrict__ out, long n) {
  for (long i = (long)blockIdx.x * blockDim.x + threadIdx.x; i < n;
       i += (long)gridDim.x * blockDim.x)
    out[i] = (__bf16)in[i];
}

// w_out (NH,NE,DHID,DH) f32 -> woutT (NH,NE,DH,DHID) bf16
__global__ void transpose_cast_wout_kernel(const float* __restrict__ in,
                                           __bf16* __restrict__ out, long n) {
  const long plane = (long)DHID * DHH;
  for (long i = (long)blockIdx.x * blockDim.x + threadIdx.x; i < n;
       i += (long)gridDim.x * blockDim.x) {
    long ne  = i / plane;
    long rem = i - ne * plane;
    long f = rem / DHH, d = rem - f * DHH;
    out[ne * plane + d * (long)DHID + f] = (__bf16)in[i];
  }
}

// ---------------- GEMM: C(MxN) = A(MxK) * W(NxK)^T + bias -------------------
// Block = 128 rows x 128 cols, 8 waves; wave w owns 2 row tiles ((w&3)*2..+1)
// x 4 col tiles ((w>>2)*4..+3) = 8 accumulators.  K in 128-wide chunks: TDM
// double-buffered into LDS; per k-step 8 WMMAs from 12 ds_load_b128 with the
// next step's loads issued before the current step's WMMAs.
__global__ __launch_bounds__(256) void gemm_nt_kernel(
    const __bf16* __restrict__ A, const __bf16* __restrict__ W,
    const float* __restrict__ bias, float* __restrict__ outF,
    __bf16* __restrict__ outB, int K, int lda, int ldw, int ldo) {
  __shared__ __bf16 ldsA[2][128 * 128];
  __shared__ __bf16 ldsB[2][128 * 128];

  const int t0   = blockIdx.x * 128;
  const int n0   = blockIdx.y * 128;
  const int lane = threadIdx.x & 31;
  const int w    = threadIdx.x >> 5;
  const int rt0  = (w & 3) * 2;   // row tiles rt0, rt0+1
  const int ct0  = (w >> 2) * 4;  // col tiles ct0..+3
  const int nch  = K / 128;

  v8f z = {};
  v8f acc[2][4] = {{z, z, z, z}, {z, z, z, z}};

  if (w == 0) {  // prologue: chunk 0 in flight
    tdm_load_2d_bf16(lds_offset_of(&ldsA[0][0]), A + (size_t)t0 * lda, 128, 128,
                     (unsigned)lda);
    tdm_load_2d_bf16(lds_offset_of(&ldsB[0][0]), W + (size_t)n0 * ldw, 128, 128,
                     (unsigned)ldw);
  }

  for (int kc = 0; kc < nch; ++kc) {
    const int b = kc & 1;
    if (w == 0) {
      if (kc + 1 < nch) {  // prefetch next chunk into other buffer
        tdm_load_2d_bf16(lds_offset_of(&ldsA[1 - b][0]),
                         A + (size_t)t0 * lda + (kc + 1) * 128, 128, 128,
                         (unsigned)lda);
        tdm_load_2d_bf16(lds_offset_of(&ldsB[1 - b][0]),
                         W + (size_t)n0 * ldw + (kc + 1) * 128, 128, 128,
                         (unsigned)ldw);
        __builtin_amdgcn_s_wait_tensorcnt(2);  // chunk kc resident (in-order)
      } else {
        __builtin_amdgcn_s_wait_tensorcnt(0);
      }
    }
    __syncthreads();  // publish chunk kc to all waves

    // register-pipelined k-steps over the chunk
    v16bf af_p[2], bf_p[4];
#pragma unroll
    for (int r = 0; r < 2; ++r)
      af_p[r] = load_a_frag(&ldsA[b][0] + (rt0 + r) * 16 * 128, 128, lane);
#pragma unroll
    for (int j = 0; j < 4; ++j)
      bf_p[j] = load_b_frag_wt(&ldsB[b][0] + (ct0 + j) * 16 * 128, 128, lane);
#pragma unroll
    for (int ksi = 0; ksi < 4; ++ksi) {
      v16bf af_n[2] = {af_p[0], af_p[1]};
      v16bf bf_n[4] = {bf_p[0], bf_p[1], bf_p[2], bf_p[3]};
      if (ksi < 3) {  // issue next step's loads before this step's WMMAs
        const int ks = (ksi + 1) * 32;
#pragma unroll
        for (int r = 0; r < 2; ++r)
          af_n[r] =
              load_a_frag(&ldsA[b][0] + (rt0 + r) * 16 * 128 + ks, 128, lane);
#pragma unroll
        for (int j = 0; j < 4; ++j)
          bf_n[j] = load_b_frag_wt(&ldsB[b][0] + (ct0 + j) * 16 * 128 + ks, 128,
                                   lane);
      }
#pragma unroll
      for (int r = 0; r < 2; ++r)
#pragma unroll
        for (int j = 0; j < 4; ++j)
          acc[r][j] = wmma_bf16(af_p[r], bf_p[j], acc[r][j]);
#pragma unroll
      for (int r = 0; r < 2; ++r) af_p[r] = af_n[r];
#pragma unroll
      for (int j = 0; j < 4; ++j) bf_p[j] = bf_n[j];
    }
    __syncthreads();  // all reads of buf b done before it is refilled
  }

  const int m0 = (lane >> 4) * 8;
  const int nl = lane & 15;
#pragma unroll
  for (int r = 0; r < 2; ++r)
#pragma unroll
    for (int j = 0; j < 4; ++j)
#pragma unroll
      for (int i = 0; i < 8; ++i) {
        int row = t0 + (rt0 + r) * 16 + m0 + i;
        int col = n0 + (ct0 + j) * 16 + nl;
        float v = acc[r][j][i] + (bias ? bias[col] : 0.0f);
        if (outF) outF[(size_t)row * ldo + col] = v;
        if (outB) outB[(size_t)row * ldo + col] = (__bf16)v;
      }
}

// ---------------- router: fp32 logits, top-2, softmax -> dense gate ---------
__global__ void router_gate_kernel(const float* __restrict__ h,
                                   const float* __restrict__ rw,
                                   float* __restrict__ gate) {
  int idx = blockIdx.x * blockDim.x + threadIdx.x;  // t*NH + n
  if (idx >= T_TOK * NHH) return;
  int t = idx / NHH, n = idx % NHH;
  const float* hp = h + (size_t)t * DM + n * DHH;
  float logits[NEXP];
#pragma unroll
  for (int e = 0; e < NEXP; ++e) {
    const float* wp = rw + ((size_t)n * NEXP + e) * DHH;
    float s = 0.0f;
    for (int d = 0; d < DHH; ++d) s = fmaf(hp[d], wp[d], s);
    logits[e] = s;
  }
  int i0 = 0;
#pragma unroll
  for (int e = 1; e < NEXP; ++e) if (logits[e] > logits[i0]) i0 = e;
  int i1 = (i0 == 0) ? 1 : 0;
#pragma unroll
  for (int e = 0; e < NEXP; ++e)
    if (e != i0 && logits[e] > logits[i1]) i1 = e;
  float ex = expf(logits[i1] - logits[i0]);  // l0 >= l1
  float g0 = 1.0f / (1.0f + ex);
  float g1 = ex * g0;
  float* gp = gate + (size_t)idx * NEXP;
#pragma unroll
  for (int e = 0; e < NEXP; ++e) gp[e] = 0.0f;
  gp[i0] = g0;
  gp[i1] = g1;
}

// ---------------- fused expert FFN ------------------------------------------
// Block = (64 tokens, head n).  H tile arrives by TDM and its A-fragments are
// hoisted into registers once (loop-invariant).  Flat loop over (expert e,
// 64-wide DHID chunk c): W_in / W_out^T chunks via double-buffered TDM;
// phase A: hid(64x64) WMMA + exact-erf GELU -> bf16 LDS; phase B: ye(64x128)
// WMMA accumulate; per expert y += gate*ye.  y stored bf16.
__global__ __launch_bounds__(256) void expert_ffn_kernel(
    const __bf16* __restrict__ h_bf, const float* __restrict__ gate,
    const __bf16* __restrict__ win, const __bf16* __restrict__ woutT,
    __bf16* __restrict__ y_bf) {
  __shared__ __bf16 lds_h[64 * 128];      // 16 KB
  __shared__ __bf16 lds_hid[64 * 64];     // 8 KB
  __shared__ float  lds_gate[64 * 8];     // 2 KB
  __shared__ __bf16 lds_wi[2][64 * 128];  // 2x16 KB  (f_local x d)
  __shared__ __bf16 lds_wo[2][128 * 64];  // 2x16 KB  (d x f_local)

  const int t0   = blockIdx.x * 64;
  const int n    = blockIdx.y;
  const int tid  = threadIdx.x;
  const int lane = tid & 31;
  const int w    = tid >> 5;

  const __bf16* WiBase = win + ((size_t)n * NEXP) * DHID * DHH;
  const __bf16* WoBase = woutT + ((size_t)n * NEXP) * DHH * DHID;

  if (w == 0) {  // prologue: H tile + (e=0,c=0) chunks in flight
    tdm_load_2d_bf16(lds_offset_of(lds_h),
                     h_bf + (size_t)t0 * DM + n * DHH, 128, 64, DM);
    tdm_load_2d_bf16(lds_offset_of(&lds_wi[0][0]), WiBase, 128, 64, 128);
    tdm_load_2d_bf16(lds_offset_of(&lds_wo[0][0]), WoBase, 64, 128, 512);
    __builtin_amdgcn_s_wait_tensorcnt(2);  // H resident (issued first)
  }
  for (int u = tid; u < 64 * 8; u += 256)
    lds_gate[u] = gate[((size_t)(t0 + (u >> 3)) * NHH + n) * NEXP + (u & 7)];
  __syncthreads();  // H + gates visible to all waves

  // wave roles
  const int ctA  = w & 3;          // phase A col tile (of 4)
  const int rtA0 = (w >> 2) * 2;   // phase A row tiles rtA0, rtA0+1
  const int rtB  = w >> 1;         // phase B row tile
  const int ctB0 = (w & 1) * 4;    // phase B col tiles ctB0..+3
  const int m0   = (lane >> 4) * 8;
  const int nl   = lane & 15;

  // hoist loop-invariant phase-A A-fragments (H tile) into registers
  v16bf haf[2][4];
#pragma unroll
  for (int r = 0; r < 2; ++r)
#pragma unroll
    for (int ksi = 0; ksi < 4; ++ksi)
      haf[r][ksi] =
          load_a_frag(lds_h + (rtA0 + r) * 16 * 128 + ksi * 32, 128, lane);

  v8f z = {};
  v8f yacc[4] = {z, z, z, z};

  for (int e = 0; e < NEXP; ++e) {
    v8f ye[4] = {z, z, z, z};
    for (int c = 0; c < 8; ++c) {
      const int it = e * 8 + c;
      const int b  = it & 1;
      if (w == 0) {
        if (it + 1 < 64) {  // prefetch next (e,c) chunks
          const int en = (it + 1) >> 3, cn = (it + 1) & 7;
          tdm_load_2d_bf16(lds_offset_of(&lds_wi[1 - b][0]),
                           WiBase + ((size_t)en * DHID + cn * 64) * DHH, 128,
                           64, 128);
          tdm_load_2d_bf16(lds_offset_of(&lds_wo[1 - b][0]),
                           WoBase + (size_t)en * DHH * DHID + cn * 64, 64, 128,
                           512);
          __builtin_amdgcn_s_wait_tensorcnt(2);  // chunk `it` resident
        } else {
          __builtin_amdgcn_s_wait_tensorcnt(0);
        }
      }
      __syncthreads();  // publish chunk to all waves

      // ---- phase A: hid(64x64) = H(64x128) @ Wi_chunk^T (B pipelined) ----
      v8f hacc[2] = {z, z};
      {
        v16bf bf_p =
            load_b_frag_wt(&lds_wi[b][0] + ctA * 16 * 128, 128, lane);
#pragma unroll
        for (int ksi = 0; ksi < 4; ++ksi) {
          v16bf bf_n = bf_p;
          if (ksi < 3)
            bf_n = load_b_frag_wt(
                &lds_wi[b][0] + ctA * 16 * 128 + (ksi + 1) * 32, 128, lane);
          hacc[0] = wmma_bf16(haf[0][ksi], bf_p, hacc[0]);
          hacc[1] = wmma_bf16(haf[1][ksi], bf_p, hacc[1]);
          bf_p = bf_n;
        }
      }
      __syncthreads();  // prior phase-B reads of lds_hid done
      {                 // exact-erf GELU -> bf16 hid chunk
        int col = ctA * 16 + nl;
#pragma unroll
        for (int r = 0; r < 2; ++r)
#pragma unroll
          for (int i = 0; i < 8; ++i) {
            float v = hacc[r][i];
            float g = 0.5f * v * (1.0f + erff(v * 0.70710678118654752f));
            lds_hid[((rtA0 + r) * 16 + m0 + i) * 64 + col] = (__bf16)g;
          }
      }
      __syncthreads();

      // ---- phase B: ye(64x128) += hid_chunk(64x64) @ Wo_chunk (pipelined) --
      {
        v16bf af_p = load_a_frag(lds_hid + rtB * 16 * 64, 64, lane);
        v16bf bf_p[4];
#pragma unroll
        for (int j = 0; j < 4; ++j)
          bf_p[j] =
              load_b_frag_wt(&lds_wo[b][0] + (ctB0 + j) * 16 * 64, 64, lane);
#pragma unroll
        for (int ksi = 0; ksi < 2; ++ksi) {
          v16bf af_n = af_p;
          v16bf bf_n[4] = {bf_p[0], bf_p[1], bf_p[2], bf_p[3]};
          if (ksi < 1) {
            af_n = load_a_frag(lds_hid + rtB * 16 * 64 + 32, 64, lane);
#pragma unroll
            for (int j = 0; j < 4; ++j)
              bf_n[j] = load_b_frag_wt(
                  &lds_wo[b][0] + (ctB0 + j) * 16 * 64 + 32, 64, lane);
          }
#pragma unroll
          for (int j = 0; j < 4; ++j) ye[j] = wmma_bf16(af_p, bf_p[j], ye[j]);
          af_p = af_n;
#pragma unroll
          for (int j = 0; j < 4; ++j) bf_p[j] = bf_n[j];
        }
      }
      __syncthreads();  // reads of buf b done before it is refilled
    }
    // gate-weighted accumulate into y
#pragma unroll
    for (int j = 0; j < 4; ++j)
#pragma unroll
      for (int i = 0; i < 8; ++i) {
        float g = lds_gate[(rtB * 16 + m0 + i) * 8 + e];
        yacc[j][i] += g * ye[j][i];
      }
  }

  // store y as bf16 (input to out-projection GEMM)
#pragma unroll
  for (int j = 0; j < 4; ++j)
#pragma unroll
    for (int i = 0; i < 8; ++i) {
      int row = t0 + rtB * 16 + m0 + i;
      int col = n * DHH + (ctB0 + j) * 16 + nl;
      y_bf[(size_t)row * DM + col] = (__bf16)yacc[j][i];
    }
}

// ---------------- launch -----------------------------------------------------
extern "C" void kernel_launch(void* const* d_in, const int* in_sizes, int n_in,
                              void* d_out, int out_size, void* d_ws,
                              size_t ws_size, hipStream_t stream) {
  const float* x        = (const float*)d_in[0];
  const float* in_w     = (const float*)d_in[1];
  const float* in_b     = (const float*)d_in[2];
  const float* router_w = (const float*)d_in[3];
  const float* w_in     = (const float*)d_in[4];
  const float* w_out    = (const float*)d_in[5];
  const float* out_w    = (const float*)d_in[6];
  const float* out_b    = (const float*)d_in[7];
  float* out = (float*)d_out;

  char* ws = (char*)d_ws;
  size_t off = 0;
  auto take = [&](size_t bytes) {
    char* p = ws + off;
    off += (bytes + 255) & ~(size_t)255;
    return p;
  };
  const long nXB  = (long)T_TOK * DM;
  const long nWin = (long)NHH * NEXP * DHID * DHH;
  __bf16* x_bf    = (__bf16*)take(nXB * 2);
  __bf16* inw_bf  = (__bf16*)take((long)DM * DM * 2);
  __bf16* win_bf  = (__bf16*)take(nWin * 2);
  __bf16* woutT   = (__bf16*)take(nWin * 2);
  __bf16* outw_bf = (__bf16*)take((long)DM * DM * 2);
  float*  h_f32   = (float*)take(nXB * 4);
  __bf16* h_bf    = (__bf16*)take(nXB * 2);
  float*  gate    = (float*)take((long)T_TOK * NHH * NEXP * 4);
  __bf16* y_bf    = (__bf16*)take(nXB * 2);

  // casts (weights + activations -> bf16)
  cast_f32_bf16_kernel<<<2048, 256, 0, stream>>>(x, x_bf, nXB);
  cast_f32_bf16_kernel<<<1024, 256, 0, stream>>>(in_w, inw_bf, (long)DM * DM);
  cast_f32_bf16_kernel<<<2048, 256, 0, stream>>>(w_in, win_bf, nWin);
  transpose_cast_wout_kernel<<<2048, 256, 0, stream>>>(w_out, woutT, nWin);
  cast_f32_bf16_kernel<<<1024, 256, 0, stream>>>(out_w, outw_bf, (long)DM * DM);

  dim3 ggrid(T_TOK / 128, DM / 128);
  // h = x @ in_w^T + in_b  (f32 for router, bf16 for experts)
  gemm_nt_kernel<<<ggrid, 256, 0, stream>>>(x_bf, inw_bf, in_b, h_f32, h_bf,
                                            DM, DM, DM, DM);
  // router -> dense gate
  router_gate_kernel<<<(T_TOK * NHH + 255) / 256, 256, 0, stream>>>(
      h_f32, router_w, gate);
  // fused expert FFN (dense over experts, gate-weighted combine)
  expert_ffn_kernel<<<dim3(T_TOK / 64, NHH), 256, 0, stream>>>(
      h_bf, gate, win_bf, woutT, y_bf);
  // out = y @ out_w^T + out_b
  gemm_nt_kernel<<<ggrid, 256, 0, stream>>>(y_bf, outw_bf, out_b, out, nullptr,
                                            DM, DM, DM, DM);
}
